// Overscan_5634997093291
// MI455X (gfx1250) — matrix operand order repaired and checked by reading, not compiled, not run on hardware
//
#include <hip/hip_runtime.h>

// ---------------------------------------------------------------------------
// MI455X (gfx1250) implementation. All GEMM-shaped work goes through
// v_wmma_f32_16x16x32_f16 (f16 operands, f32 accumulate). wave32, 8 waves/WG.
// ---------------------------------------------------------------------------

typedef __attribute__((ext_vector_type(16))) _Float16 v16h;
typedef __attribute__((ext_vector_type(8)))  float    v8f;

// A fragment (16x32 f16) from a row-major [M x K] image at (row0,k0) (p pre-offset).
// lane<16 row=lane, elems = K0..15; lane>=16 row=lane-16, elems = K16..31.
__device__ __forceinline__ v16h frag_a(const _Float16* p, int ld) {
  int l = threadIdx.x & 31;
  const _Float16* b = p + (l & 15) * ld + ((l >> 4) << 4);
  v16h f;
#pragma unroll
  for (int e = 0; e < 16; ++e) f[e] = b[e];
  return f;
}

// B fragment (32x16 f16) gathered from Bt = row-major [N x K] (B transposed):
// lane<16 col=lane, elems = K0..15; lane>=16 col=lane-16, elems = K16..31.
__device__ __forceinline__ v16h frag_bt(const _Float16* p, int ld) {
  int l = threadIdx.x & 31;
  const _Float16* b = p + (l & 15) * ld + ((l >> 4) << 4);
  v16h f;
#pragma unroll
  for (int e = 0; e < 16; ++e) f[e] = b[e];
  return f;
}

__device__ __forceinline__ v8f wmma_f16(v16h a, v16h b, v8f c) {
  return __builtin_amdgcn_wmma_f32_16x16x32_f16(false, a, false, b, (short)0, c,
                                                false, false);
}

// ---------------------------------------------------------------------------
__global__ void k_f32_to_f16(const float* __restrict__ s, _Float16* __restrict__ d, int n) {
  int i = blockIdx.x * 256 + threadIdx.x;
  if (i < n) d[i] = (_Float16)s[i];
}

// pixel_unshuffle(x,2) + 1x1 conv 12->64 : (8,3,400,400) -> (8,64,200,200)
__global__ void k_unshuffle_proj(const float* __restrict__ x, const float* __restrict__ pw,
                                 const float* __restrict__ pb, float* __restrict__ y) {
  int i = blockIdx.x * 256 + threadIdx.x;
  if (i >= 20480000) return;
  int w = i % 200; int t = i / 200;
  int h = t % 200; t /= 200;
  int c = t % 64;  int b = t / 64;
  float acc = pb[c];
#pragma unroll
  for (int ci = 0; ci < 3; ++ci)
#pragma unroll
    for (int r1 = 0; r1 < 2; ++r1)
#pragma unroll
      for (int r2 = 0; r2 < 2; ++r2)
        acc += pw[c * 12 + ci * 4 + r1 * 2 + r2] *
               x[(((size_t)b * 3 + ci) * 400 + (2 * h + r1)) * 400 + (2 * w + r2)];
  y[i] = acc;
}

// Fused window MHA: one workgroup per window, tokens padded to 64.
template <int WS, int NTOK>
__global__ __launch_bounds__(256) void k_window_mha(
    const float* x, float* y,                                    // may alias (in-place)
    const _Float16* __restrict__ w_inT, const float* __restrict__ b_in,
    const _Float16* __restrict__ w_outT, const float* __restrict__ b_out,
    const float* __restrict__ bias_table, const int* __restrict__ idx,
    int H, int W, int nh, int nw) {
  __shared__ __align__(16) unsigned char smem[49152];
  _Float16* sX  = (_Float16*)smem;             // [A,B]   64 tok x 64 ch
  _Float16* sQK = (_Float16*)(smem + 8192);    // [B,C]   64 x (q0..63 | k64..127)
  _Float16* sVt = (_Float16*)(smem + 24576);   // [B,E]   64 ch x 64 tok (V^T)
  _Float16* sS  = (_Float16*)(smem + 32768);   // [C,D]   2 x 64 x 64 scores
  _Float16* sP  = (_Float16*)smem;             // [D,E]   2 x 64 x 64 probs (overlay)
  _Float16* sO  = (_Float16*)(smem + 32768);   // [E,F]   64 x 64 (overlay)

  const int wave = threadIdx.x >> 5;
  const int lane = threadIdx.x & 31;
  int ww = blockIdx.x % nw;
  int t0 = blockIdx.x / nw;
  int wh = t0 % nh;
  int b  = t0 / nh;
  const size_t plane = (size_t)H * W;
  const float* xb = x + (size_t)b * 64 * plane;
  float* yb = y + (size_t)b * 64 * plane;

  // Phase A: stage window tokens (zero-pad to 64 tokens), f32 -> f16
  for (int i = threadIdx.x; i < 64 * 64; i += 256) {
    int n = i >> 6, c = i & 63;
    float v = 0.f;
    if (n < NTOK) {
      int hh = wh * WS + n / WS;
      int wp = ww * WS + n % WS;
      v = xb[c * plane + (size_t)hh * W + wp];
    }
    sX[i] = (_Float16)v;
  }
  __syncthreads();

  // Phase B: QKV = X @ w_in^T + b_in   (64x64 @ 64x192), q pre-scaled, V transposed
  const float scale = 0.17677669529663687f;  // 32^-0.5
  for (int t = wave; t < 48; t += 8) {
    int mt = t / 12, nt = t % 12;
    v8f acc = {};
#pragma unroll
    for (int kk = 0; kk < 64; kk += 32) {
      v16h a  = frag_a (sX + mt * 16 * 64 + kk, 64);
      v16h bb = frag_bt(w_inT + nt * 16 * 64 + kk, 64);
      acc = wmma_f16(a, bb, acc);
    }
    int col = nt * 16 + (lane & 15);
    float bias = b_in[col];
    int rb = (lane >> 4) * 8;
#pragma unroll
    for (int r = 0; r < 8; ++r) {
      int row = mt * 16 + rb + r;
      float v = acc[r] + bias;
      if (col < 64)       sQK[row * 128 + col] = (_Float16)(v * scale);
      else if (col < 128) sQK[row * 128 + col] = (_Float16)v;
      else                sVt[(col - 128) * 64 + row] = (_Float16)v;
    }
  }
  __syncthreads();

  // Phase C: per-head scores S = q @ k^T + rel-pos bias  (2 heads, hd=32)
  for (int t = wave; t < 32; t += 8) {
    int hD = t >> 4; int tt = t & 15;
    int nt = tt >> 2, mt = tt & 3;  // nt: query tile, mt: key tile
    v16h a  = frag_a (sQK + nt * 16 * 128 + hD * 32, 128);
    v16h bb = frag_bt(sQK + mt * 16 * 128 + 64 + hD * 32, 128);
    v8f acc = {};
    acc = wmma_f16(a, bb, acc);
    int col = mt * 16 + (lane & 15);
    int rb = (lane >> 4) * 8;
#pragma unroll
    for (int r = 0; r < 8; ++r) {
      int row = nt * 16 + rb + r;
      float v = acc[r];
      if (row < NTOK && col < NTOK) v += bias_table[idx[row * NTOK + col]];
      sS[(hD * 64 + row) * 64 + col] = (_Float16)v;
    }
  }
  __syncthreads();

  // Phase D: row softmax (padded key columns masked to prob 0)
  for (int rr = threadIdx.x; rr < 128; rr += 256) {
    const _Float16* Sr = sS + rr * 64;
    _Float16* Pr = sP + rr * 64;
    int n = rr & 63;
    if (n < NTOK) {
      float mx = -1e30f;
      for (int m = 0; m < NTOK; ++m) mx = fmaxf(mx, (float)Sr[m]);
      float sum = 0.f;
      for (int m = 0; m < NTOK; ++m) sum += __expf((float)Sr[m] - mx);
      float inv = 1.f / sum;
      for (int m = 0; m < 64; ++m)
        Pr[m] = (m < NTOK) ? (_Float16)(__expf((float)Sr[m] - mx) * inv) : (_Float16)0.f;
    } else {
      for (int m = 0; m < 64; ++m) Pr[m] = (_Float16)0.f;
    }
  }
  __syncthreads();

  // Phase E: O = P @ V  (per head 64x64 @ 64x32, V supplied transposed)
  for (int t = wave; t < 16; t += 8) {
    int hD = t >> 3; int tt = t & 7;
    int mt = tt >> 1, nt = tt & 1;
    v8f acc = {};
#pragma unroll
    for (int kk = 0; kk < 64; kk += 32) {
      v16h a  = frag_a (sP + (hD * 64 + mt * 16) * 64 + kk, 64);
      v16h bb = frag_bt(sVt + (hD * 32 + nt * 16) * 64 + kk, 64);
      acc = wmma_f16(a, bb, acc);
    }
    int col = hD * 32 + nt * 16 + (lane & 15);
    int rb = (lane >> 4) * 8;
#pragma unroll
    for (int r = 0; r < 8; ++r)
      sO[(mt * 16 + rb + r) * 64 + col] = (_Float16)acc[r];
  }
  __syncthreads();

  // Phase F: Y = O @ w_out^T + b_out + residual (write real tokens only)
  for (int t = wave; t < 16; t += 8) {
    int mt = t >> 2, nt = t & 3;
    v8f acc = {};
#pragma unroll
    for (int kk = 0; kk < 64; kk += 32) {
      v16h a  = frag_a (sO + mt * 16 * 64 + kk, 64);
      v16h bb = frag_bt(w_outT + nt * 16 * 64 + kk, 64);
      acc = wmma_f16(a, bb, acc);
    }
    int col = nt * 16 + (lane & 15);
    float bo = b_out[col];
    int rb = (lane >> 4) * 8;
#pragma unroll
    for (int r = 0; r < 8; ++r) {
      int row = mt * 16 + rb + r;
      if (row < NTOK) {
        int hh = wh * WS + row / WS;
        int wp = ww * WS + row % WS;
        size_t gi = col * plane + (size_t)hh * W + wp;
        yb[gi] = xb[gi] + acc[r] + bo;
      }
    }
  }
}

// 1x1 conv 64->64 + leaky ReLU as WMMA GEMM over 64-pixel tiles.
__global__ __launch_bounds__(256) void k_conv1x1_lrelu(
    const float* __restrict__ x, float* __restrict__ y,
    const _Float16* __restrict__ wA, const float* __restrict__ bias) {
  __shared__ __align__(16) _Float16 sXt[64 * 64];  // [pixel][channel]
  int wave = threadIdx.x >> 5, lane = threadIdx.x & 31;
  int b = blockIdx.x / 625;
  int p0 = (blockIdx.x % 625) * 64;
  const float* xb = x + (size_t)b * 64 * 40000 + p0;
  for (int i = threadIdx.x; i < 4096; i += 256) {
    int c = i >> 6, p = i & 63;  // coalesced over p for fixed c
    sXt[p * 64 + c] = (_Float16)xb[(size_t)c * 40000 + p];
  }
  __syncthreads();
  for (int t = wave; t < 16; t += 8) {
    int mt = t >> 2, nt = t & 3;
    v8f acc = {};
#pragma unroll
    for (int kk = 0; kk < 64; kk += 32) {
      v16h a  = frag_a (wA + mt * 16 * 64 + kk, 64);
      v16h bb = frag_bt(sXt + nt * 16 * 64 + kk, 64);
      acc = wmma_f16(a, bb, acc);
    }
    int p = nt * 16 + (lane & 15);
    int rb = (lane >> 4) * 8;
#pragma unroll
    for (int r = 0; r < 8; ++r) {
      int o = mt * 16 + rb + r;
      float v = acc[r] + bias[o];
      v = v > 0.f ? v : 0.1f * v;
      y[((size_t)b * 64 + o) * 40000 + p0 + p] = v;
    }
  }
}

// Valid 3x3 conv 64->64 (implicit GEMM, K=576) + leaky ReLU + cropped residual.
__global__ __launch_bounds__(256) void k_conv3x3_lrelu_add(
    const float* __restrict__ hin, const float* __restrict__ xres,
    float* __restrict__ y, const _Float16* __restrict__ wA,
    const float* __restrict__ bias) {
  __shared__ __align__(16) _Float16 sIn[64 * 3 * 66];  // [ch][ky][col]
  int wave = threadIdx.x >> 5, lane = threadIdx.x & 31;
  int wt = blockIdx.x & 3;
  int t0 = blockIdx.x >> 2;
  int h = t0 % 198;
  int b = t0 / 198;
  int w0 = wt * 64;
  for (int i = threadIdx.x; i < 64 * 3 * 66; i += 256) {
    int c = i / 198;
    int rem = i - c * 198;
    int ky = rem / 66, cx = rem - ky * 66;
    int gw = w0 + cx;
    float v = 0.f;
    if (gw < 200) v = hin[((size_t)b * 64 + c) * 40000 + (size_t)(h + ky) * 200 + gw];
    sIn[i] = (_Float16)v;
  }
  __syncthreads();
  for (int t = wave; t < 16; t += 8) {
    int mt = t >> 2, nt = t & 3;
    v8f acc = {};
    for (int ks = 0; ks < 18; ++ks) {  // K = 576 = 18 x 32
      v16h a = frag_a(wA + mt * 16 * 576 + ks * 32, 576);
      int n = nt * 16 + (lane & 15);
      int kb = ks * 32 + ((lane >> 4) << 4);
      v16h bb;
#pragma unroll
      for (int e = 0; e < 16; ++e) {  // im2col gather: k = ci*9 + ky*3 + kx
        int k = kb + e;
        int c = k / 9;
        int r9 = k - c * 9;
        int ky = r9 / 3, kx = r9 - ky * 3;
        bb[e] = sIn[c * 198 + ky * 66 + n + kx];
      }
      acc = wmma_f16(a, bb, acc);
    }
    int p = nt * 16 + (lane & 15);
    int rb = (lane >> 4) * 8;
#pragma unroll
    for (int r = 0; r < 8; ++r) {
      int o = mt * 16 + rb + r;
      if (w0 + p < 198) {
        float v = acc[r] + bias[o];
        v = v > 0.f ? v : 0.1f * v;
        y[((size_t)b * 64 + o) * 39204 + (size_t)h * 198 + (w0 + p)] =
            v + xres[((size_t)b * 64 + o) * 40000 + (size_t)(h + 1) * 200 + (w0 + p + 1)];
      }
    }
  }
}

// pixel_shuffle(x,2): (8,64,198,198) -> (8,16,396,396)
__global__ void k_pixel_shuffle(const float* __restrict__ x, float* __restrict__ out) {
  int i = blockIdx.x * 256 + threadIdx.x;
  if (i >= 20072448) return;
  int wo = i % 396; int t = i / 396;
  int ho = t % 396; t /= 396;
  int co = t % 16;  int b = t / 16;
  int ci = co * 4 + (ho & 1) * 2 + (wo & 1);
  out[i] = x[((size_t)b * 64 + ci) * 39204 + (size_t)(ho >> 1) * 198 + (wo >> 1)];
}

// ---------------------------------------------------------------------------
extern "C" void kernel_launch(void* const* d_in, const int* in_sizes, int n_in,
                              void* d_out, int out_size, void* d_ws, size_t ws_size,
                              hipStream_t stream) {
  (void)in_sizes; (void)n_in; (void)out_size; (void)ws_size;
  const float* x      = (const float*)d_in[0];
  const float* proj_w = (const float*)d_in[1];
  const float* proj_b = (const float*)d_in[2];
  const float* m1iw   = (const float*)d_in[3];
  const float* m1ib   = (const float*)d_in[4];
  const float* m1ow   = (const float*)d_in[5];
  const float* m1ob   = (const float*)d_in[6];
  const float* m2iw   = (const float*)d_in[7];
  const float* m2ib   = (const float*)d_in[8];
  const float* m2ow   = (const float*)d_in[9];
  const float* m2ob   = (const float*)d_in[10];
  const float* bt1    = (const float*)d_in[11];
  const float* bt2    = (const float*)d_in[12];
  const float* w1     = (const float*)d_in[13];
  const float* b1     = (const float*)d_in[14];
  const float* w2     = (const float*)d_in[15];
  const float* b2     = (const float*)d_in[16];
  const int*   idx1   = (const int*)d_in[17];
  const int*   idx2   = (const int*)d_in[18];

  // Workspace layout (f32 buffers, then f16 weight copies) ~233 MB total.
  float* xbuf = (float*)d_ws;           // (8,64,200,200) x0 / x1 in-place
  float* h1   = xbuf + 20480000;        // (8,64,200,200)
  float* x2   = h1 + 20480000;          // (8,64,198,198)
  _Float16* f_m1iw = (_Float16*)(x2 + 20072448);
  _Float16* f_m1ow = f_m1iw + 12288;
  _Float16* f_m2iw = f_m1ow + 4096;
  _Float16* f_m2ow = f_m2iw + 12288;
  _Float16* f_w1   = f_m2ow + 4096;
  _Float16* f_w2   = f_w1 + 4096;

  k_f32_to_f16<<<48, 256, 0, stream>>>(m1iw, f_m1iw, 12288);
  k_f32_to_f16<<<16, 256, 0, stream>>>(m1ow, f_m1ow, 4096);
  k_f32_to_f16<<<48, 256, 0, stream>>>(m2iw, f_m2iw, 12288);
  k_f32_to_f16<<<16, 256, 0, stream>>>(m2ow, f_m2ow, 4096);
  k_f32_to_f16<<<16, 256, 0, stream>>>(w1, f_w1, 4096);
  k_f32_to_f16<<<144, 256, 0, stream>>>(w2, f_w2, 36864);

  k_unshuffle_proj<<<80000, 256, 0, stream>>>(x, proj_w, proj_b, xbuf);

  // MHA1: 8 * 25 * 25 windows, in-place residual
  k_window_mha<8, 64><<<5000, 256, 0, stream>>>(xbuf, xbuf, f_m1iw, m1ib, f_m1ow,
                                                m1ob, bt1, idx1, 200, 200, 25, 25);

  k_conv1x1_lrelu<<<5000, 256, 0, stream>>>(xbuf, h1, f_w1, b1);
  k_conv3x3_lrelu_add<<<6336, 256, 0, stream>>>(h1, xbuf, x2, f_w2, b2);

  // MHA2: 8 * 33 * 33 windows, tokens 36 padded to 64, in-place residual
  k_window_mha<6, 36><<<8712, 256, 0, stream>>>(x2, x2, f_m2iw, m2ib, f_m2ow,
                                                m2ob, bt2, idx2, 198, 198, 33, 33);

  k_pixel_shuffle<<<78408, 256, 0, stream>>>(x2, (float*)d_out);
}
